// DynamicMoELayer_84928683311855
// MI455X (gfx1250) — matrix operand order repaired and checked by reading, not compiled, not run on hardware
//
#include <hip/hip_runtime.h>
#include <math.h>

// ---------------------------------------------------------------------------
// DynamicMoELayer fused kernel for MI455X (gfx1250, wave32, WMMA).
// Shapes: B=4 T=2048 C=512 I=1024 E=8, N = B*T = 8192.
// Strategy:
//   k1: routing weights (fp32, one wave per token) -> ws.rw [N][8]
//   k2: fp32 -> bf16 conversions (hidden as-is, w1 transposed to [E][I][C],
//       w2 transposed to [E][C][I]) so WMMA B columns are contiguous.
//   k3: fused expert FFN. Per block: 16 tokens x full C=512 output.
//       stage1: inter = gelu(h @ w1[e]) * rw[:,e]  (bf16 into LDS)
//       stage2: acc  += inter @ w2[e]              (fp32 WMMA accum chain)
// Workspace: 256KB + 3 * 8,388,608 B ~= 25.4 MB.
// ---------------------------------------------------------------------------

typedef __bf16 bf16;
typedef bf16  v16bf __attribute__((ext_vector_type(16)));
typedef float v8f   __attribute__((ext_vector_type(8)));

static __device__ __forceinline__ v16bf ld16(const bf16* p) {
  union { uint4 q[2]; v16bf v; } u;
  u.q[0] = *(const uint4*)(p);
  u.q[1] = *(const uint4*)(p + 8);
  return u.v;
}

// A-fragment: two non-adjacent K-octets from a row (row-major, bf16).
static __device__ __forceinline__ v16bf ld_a(const bf16* rowp, int o1) {
  union { uint4 q[2]; v16bf v; } u;
  u.q[0] = *(const uint4*)(rowp + o1);
  u.q[1] = *(const uint4*)(rowp + o1 + 16);
  return u.v;
}

// ---------------------------------------------------------------------------
// Kernel 1: routing weights. One wave (32 lanes) per token.
// ---------------------------------------------------------------------------
__global__ __launch_bounds__(256)
void routing_kernel(const float* __restrict__ hidden,   // [N][512]
                    const float* __restrict__ sim,      // [512][8]
                    const float* __restrict__ gates,    // [8]
                    const int*   __restrict__ min_experts,
                    float* __restrict__ rw) {           // [N][8]
  __shared__ __align__(16) float s_sim[512 * 8];
  __shared__ float s_colss[8];
  __shared__ float s_inv[8];
  __shared__ float s_sig[8];
  const int tid = threadIdx.x;

  for (int i = tid; i < 512 * 8; i += 256) s_sim[i] = sim[i];
  if (tid < 8) s_colss[tid] = 0.f;
  __syncthreads();

  // per-expert column sum of squares
  {
    float part = 0.f;
    const int e = tid & 7;
    for (int r = tid >> 3; r < 512; r += 32) {
      float v = s_sim[r * 8 + e];
      part += v * v;
    }
    atomicAdd(&s_colss[e], part);
  }
  __syncthreads();
  if (tid < 8) {
    s_inv[tid] = 1.f / fmaxf(sqrtf(s_colss[tid]), 1e-12f);
    s_sig[tid] = 1.f / (1.f + expf(-gates[tid]));
  }
  __syncthreads();

  const int w    = tid >> 5;
  const int lane = tid & 31;
  const int t    = blockIdx.x * 8 + w;
  const float* row = hidden + (size_t)t * 512;

  float ss = 0.f;
  float d[8];
#pragma unroll
  for (int j = 0; j < 8; ++j) d[j] = 0.f;

  for (int c = lane; c < 512; c += 32) {
    float h = row[c];
    ss += h * h;
    const float* sc = &s_sim[c * 8];
#pragma unroll
    for (int j = 0; j < 8; ++j) d[j] += h * sc[j];
  }
#pragma unroll
  for (int off = 16; off > 0; off >>= 1) {
    ss += __shfl_xor(ss, off, 32);
#pragma unroll
    for (int j = 0; j < 8; ++j) d[j] += __shfl_xor(d[j], off, 32);
  }

  const float invh = 1.f / fmaxf(sqrtf(ss), 1e-12f);
  float lg[8], gt[8];
  int msk = 0;
#pragma unroll
  for (int j = 0; j < 8; ++j) {
    lg[j] = d[j] * invh * s_inv[j] - s_sig[j];
    gt[j] = fmaxf(lg[j], 0.f);
    if (gt[j] > 0.f) msk |= (1 << j);
  }
  if (msk == 0) {
    int mk = *min_experts;
    if (mk > 8) mk = 8;
    for (int it = 0; it < mk; ++it) {
      float best = -INFINITY;
      int bi = 0;
#pragma unroll
      for (int j = 0; j < 8; ++j)
        if (!((msk >> j) & 1) && lg[j] > best) { best = lg[j]; bi = j; }
      msk |= (1 << bi);
    }
  }
  float mx = -INFINITY;
#pragma unroll
  for (int j = 0; j < 8; ++j)
    if ((msk >> j) & 1) mx = fmaxf(mx, gt[j]);
  float s = 0.f, wg[8];
#pragma unroll
  for (int j = 0; j < 8; ++j) {
    wg[j] = ((msk >> j) & 1) ? expf(gt[j] - mx) : 0.f;
    s += wg[j];
  }
  const float inv = 1.f / s;
  if (lane == 0) {
#pragma unroll
    for (int j = 0; j < 8; ++j) rw[(size_t)t * 8 + j] = wg[j] * inv;
  }
}

// ---------------------------------------------------------------------------
// Kernel 2: fp32 -> bf16 conversions (with transposes for weight tensors).
// ---------------------------------------------------------------------------
__global__ __launch_bounds__(256)
void cvt_hidden(const float* __restrict__ in, bf16* __restrict__ out) {
  size_t i = (size_t)blockIdx.x * 256 + threadIdx.x;  // 8192*512 elements
  out[i] = (bf16)in[i];
}

__global__ __launch_bounds__(256)
void cvt_w1(const float* __restrict__ in, bf16* __restrict__ out) {
  // in [E][C=512][I=1024] -> out [E][I][C]
  int o = blockIdx.x * 256 + threadIdx.x;   // 8*1024*512 = 2^22
  int c = o & 511;
  int i = (o >> 9) & 1023;
  int e = o >> 19;
  out[o] = (bf16)in[((size_t)(e * 512 + c)) * 1024 + i];
}

__global__ __launch_bounds__(256)
void cvt_w2(const float* __restrict__ in, bf16* __restrict__ out) {
  // in [E][I=1024][C=512] -> out [E][C][I]
  int o = blockIdx.x * 256 + threadIdx.x;
  int i = o & 1023;
  int c = (o >> 10) & 511;
  int e = o >> 19;
  out[o] = (bf16)in[((size_t)(e * 1024 + i)) * 512 + c];
}

// ---------------------------------------------------------------------------
// Kernel 3: fused expert FFN. 256 threads = 8 waves; block owns 16 tokens.
// ---------------------------------------------------------------------------
__global__ __launch_bounds__(256)
void moe_main(const bf16* __restrict__ hB,    // [N][512] bf16
              const bf16* __restrict__ w1T,   // [E][I=1024][C=512] bf16
              const bf16* __restrict__ w2T,   // [E][C=512][I=1024] bf16
              const float* __restrict__ rw,   // [N][8]
              float* __restrict__ out) {      // [N][512] fp32
  __shared__ __align__(16) bf16  sh_hidden[16 * 512];   // 16 KB
  __shared__ __align__(16) bf16  sh_inter[16 * 1024];   // 32 KB
  __shared__ float sh_rw[16 * 8];

  const int tid  = threadIdx.x;
  const int w    = tid >> 5;       // wave id 0..7
  const int lane = tid & 31;
  const int hi   = lane >> 4;      // half-wave select
  const int nl   = lane & 15;      // row (A) / column (B,D) index
  const int tok0 = blockIdx.x * 16;

  // Stage hidden tile 16x512 bf16 into LDS (1024 uint4).
  {
    const uint4* src = (const uint4*)(hB + (size_t)tok0 * 512);
    uint4* dst = (uint4*)sh_hidden;
#pragma unroll
    for (int k = 0; k < 4; ++k) dst[tid + k * 256] = src[tid + k * 256];
  }
  if (tid < 128) sh_rw[tid] = rw[(size_t)tok0 * 8 + tid];
  __syncthreads();

  const v8f vzero = {0.f, 0.f, 0.f, 0.f, 0.f, 0.f, 0.f, 0.f};
  v8f acc[4];
#pragma unroll
  for (int ct = 0; ct < 4; ++ct) acc[ct] = vzero;

#pragma unroll 1
  for (int e = 0; e < 8; ++e) {
    // routing weight for each D-row this lane produces in stage 1
    float rwv[8];
#pragma unroll
    for (int v = 0; v < 8; ++v) rwv[v] = sh_rw[(hi * 8 + v) * 8 + e];

    // ---- Stage 1: inter = gelu(hidden @ w1[e]) * rw, per-wave 8 I-tiles ----
    const bf16* arow = sh_hidden + nl * 512;
#pragma unroll 1
    for (int tt = 0; tt < 8; ++tt) {
      const int i0 = (w * 8 + tt) * 16;
      const bf16* bcol = w1T + ((size_t)(e * 1024 + i0 + nl)) * 512 + hi * 16;
      v8f c = vzero;
#pragma unroll
      for (int kt = 0; kt < 16; ++kt) {
        v16bf a = ld16(arow + kt * 32 + hi * 16);
        v16bf b = ld16(bcol + kt * 32);
        c = __builtin_amdgcn_wmma_f32_16x16x32_bf16(
                false, a, false, b, (short)0, c, false, false);
      }
#pragma unroll
      for (int v = 0; v < 8; ++v) {
        float x = c[v];
        float g = 0.5f * x * (1.f + erff(x * 0.70710678118654752f)) * rwv[v];
        sh_inter[(v + 8 * hi) * 1024 + i0 + nl] = (bf16)g;
      }
    }
    __syncthreads();

    // ---- Stage 2: acc += inter @ w2[e], per-wave 4 C-tiles, K = 1024 ----
    const bf16* arow2 = sh_inter + nl * 1024;
#pragma unroll 1
    for (int kt = 0; kt < 32; ++kt) {
      v16bf a = ld16(arow2 + kt * 32 + hi * 16);
#pragma unroll
      for (int ct = 0; ct < 4; ++ct) {
        const int n0 = (w * 4 + ct) * 16;
        v16bf b = ld16(w2T + ((size_t)(e * 512 + n0 + nl)) * 1024 +
                       kt * 32 + hi * 16);
        acc[ct] = __builtin_amdgcn_wmma_f32_16x16x32_bf16(
                      false, a, false, b, (short)0, acc[ct], false, false);
      }
    }
    __syncthreads();
  }

  // ---- Epilogue: write fp32 output tile ----
#pragma unroll
  for (int ct = 0; ct < 4; ++ct) {
    const int n = (w * 4 + ct) * 16 + nl;
#pragma unroll
    for (int v = 0; v < 8; ++v)
      out[(size_t)(tok0 + v + 8 * hi) * 512 + n] = acc[ct][v];
  }
}

// ---------------------------------------------------------------------------
// Launch
// ---------------------------------------------------------------------------
extern "C" void kernel_launch(void* const* d_in, const int* in_sizes, int n_in,
                              void* d_out, int out_size, void* d_ws, size_t ws_size,
                              hipStream_t stream) {
  const float* hidden      = (const float*)d_in[0];  // [4,2048,512]
  const float* sim         = (const float*)d_in[1];  // [512,8]
  const float* gates       = (const float*)d_in[2];  // [8]
  const float* w1          = (const float*)d_in[3];  // [8,512,1024]
  const float* w2          = (const float*)d_in[4];  // [8,1024,512]
  const int*   min_experts = (const int*)d_in[5];    // scalar
  float* out = (float*)d_out;                        // [4,2048,512]

  char* ws = (char*)d_ws;
  float* rw = (float*)ws;                                  //   262,144 B
  bf16* hB  = (bf16*)(ws + 262144);                        // 8,388,608 B
  bf16* w1T = (bf16*)(ws + 262144 + 8388608);              // 8,388,608 B
  bf16* w2T = (bf16*)(ws + 262144 + 2 * 8388608);          // 8,388,608 B

  routing_kernel<<<1024, 256, 0, stream>>>(hidden, sim, gates, min_experts, rw);
  cvt_hidden<<<16384, 256, 0, stream>>>(hidden, hB);   // 8192*512 / 256
  cvt_w1<<<16384, 256, 0, stream>>>(w1, w1T);          // 2^22 / 256
  cvt_w2<<<16384, 256, 0, stream>>>(w2, w2T);
  moe_main<<<512, 256, 0, stream>>>(hB, w1T, w2T, rw, out);
}
